// SimAMVisionMamba_18777597018404
// MI455X (gfx1250) — matrix operand order repaired and checked by reading, not compiled, not run on hardware
//
#include <hip/hip_runtime.h>

// ---------------- problem constants (reference shapes) ----------------
#define BB   2            // batch
#define DM   128          // d_model
#define DI   256          // d_inner
#define DSN  16           // d_state
#define DTR  8            // dt_rank
#define LL   4096         // H*W
#define BL   (BB*LL)      // 8192 tokens total
#define NXP  48           // x_dbl width padded 40 -> 48 (3 WMMA n-tiles)
#define CS   128          // scan chunk length
#define NC   (LL/CS)      // 32 chunks per (batch,direction)

typedef __attribute__((ext_vector_type(16))) _Float16 v16h;
typedef __attribute__((ext_vector_type(8)))  float    v8f;

static __device__ __forceinline__ float sigmoidf_(float v) {
    return 1.0f / (1.0f + __expf(-v));
}

// LDS byte offset of a generic pointer to a __shared__ object:
// flat LDS representation is {SHARED_BASE[63:32], lds_offset[31:0]} (ISA 10.2),
// so truncation to 32 bits yields the DS address.
static __device__ __forceinline__ unsigned lds_off_(const void* p) {
    return (unsigned)(unsigned long long)(uintptr_t)p;
}

// async copy: 4 bytes global -> LDS (per-lane addresses), tracked by ASYNCcnt
static __device__ __forceinline__ void async_g2l_b32(unsigned lds, const void* g) {
    unsigned long long ga = (unsigned long long)(uintptr_t)g;
    asm volatile("global_load_async_to_lds_b32 %0, %1, off"
                 :: "v"(lds), "v"(ga) : "memory");
}
// async copy: 16 bytes global -> LDS
static __device__ __forceinline__ void async_g2l_b128(unsigned lds, const void* g) {
    unsigned long long ga = (unsigned long long)(uintptr_t)g;
    asm volatile("global_load_async_to_lds_b128 %0, %1, off"
                 :: "v"(lds), "v"(ga) : "memory");
}
static __device__ __forceinline__ void wait_async0_() {
    asm volatile("s_wait_asynccnt 0x0" ::: "memory");
}

// ---------------- weight prep ----------------
__global__ __launch_bounds__(256) void cvt_f32_to_f16(const float* __restrict__ s,
                                                      _Float16* __restrict__ d, int n) {
    int i = blockIdx.x * 256 + threadIdx.x;
    if (i < n) d[i] = (_Float16)s[i];
}

// W_x is (40,256); pad to (48,256) with zero rows so every WMMA n-tile is valid
__global__ __launch_bounds__(256) void pad_wx_f16(const float* __restrict__ s,
                                                  _Float16* __restrict__ d) {
    int i = blockIdx.x * 256 + threadIdx.x;        // over 48*256
    if (i >= NXP * DI) return;
    int row = i / DI, col = i % DI;
    d[i] = (row < (DTR + 2 * DSN)) ? (_Float16)s[row * DI + col] : (_Float16)0.0f;
}

__global__ __launch_bounds__(256) void negexp_f32(const float* __restrict__ s,
                                                  float* __restrict__ d, int n) {
    int i = blockIdx.x * 256 + threadIdx.x;
    if (i < n) d[i] = -__expf(s[i]);
}

// ---------------- LayerNorm + transpose (NCHW -> token-major) + f16 ----------------
// x: (B, DM, LL). Writes t_f16[(b*LL+l)*DM + c].
// Tile staged to LDS with CDNA5 async global->LDS copies (no VGPR round-trip).
__global__ __launch_bounds__(256) void ln_kernel(const float* __restrict__ x,
                                                 const float* __restrict__ lnw,
                                                 const float* __restrict__ lnb,
                                                 _Float16* __restrict__ tbf) {
    __shared__ float tile[DM][33];                 // 32 tokens + pad (bank-conflict free)
    const int blk = blockIdx.x;                    // 0 .. B*LL/32 - 1
    const int b   = blk / (LL / 32);
    const int l0  = (blk % (LL / 32)) * 32;
    const int t   = threadIdx.x;
    #pragma unroll
    for (int it = 0; it < 16; ++it) {              // 128ch x 32tok, coalesced along l
        int idx = it * 256 + t;
        int c = idx >> 5, li = idx & 31;
        async_g2l_b32(lds_off_(&tile[c][li]),
                      x + ((size_t)(b * DM + c)) * LL + l0 + li);
    }
    wait_async0_();
    __syncthreads();
    const int wave = t >> 5, lane = t & 31;
    #pragma unroll
    for (int pass = 0; pass < 4; ++pass) {         // 8 waves x 4 passes = 32 tokens
        int li = pass * 8 + wave;
        float s = 0.f, s2 = 0.f;
        #pragma unroll
        for (int k = 0; k < 4; ++k) {
            float v = tile[lane + 32 * k][li];
            s += v; s2 += v * v;
        }
        #pragma unroll
        for (int m = 16; m >= 1; m >>= 1) {        // wave32 reduction
            s  += __shfl_xor(s,  m, 32);
            s2 += __shfl_xor(s2, m, 32);
        }
        float mu   = s * (1.0f / DM);
        float var  = s2 * (1.0f / DM) - mu * mu;
        float rstd = rsqrtf(var + 1e-5f);
        #pragma unroll
        for (int k = 0; k < 4; ++k) {
            int c = lane + 32 * k;
            float v = (tile[c][li] - mu) * rstd * lnw[c] + lnb[c];
            tbf[((size_t)(b * LL) + l0 + li) * DM + c] = (_Float16)v;
        }
    }
}

// ---------------- WMMA GEMM:  C[MxN] (+)= A[MxK](f16, row-major) * W[NxK]^T ----------------
// One wave -> one 16x16 D tile. K must be a multiple of 64, M,N multiples of 16.
__global__ __launch_bounds__(128) void wmma_gemm_f16(const _Float16* __restrict__ A,
                                                     const _Float16* __restrict__ W,
                                                     float* __restrict__ C,
                                                     int M, int N, int K, int accumulate) {
    const int lane = threadIdx.x & 31;
    const int wave = threadIdx.x >> 5;
    const int m0 = blockIdx.x << 4;
    const int n0 = (blockIdx.y * 4 + wave) << 4;
    if (n0 >= N) return;                            // whole wave exits: EXEC stays uniform
    const int nl = lane & 15;
    const int hi = lane >> 4;
    float* cp = C + (size_t)(m0 + hi * 8) * N + n0 + nl;   // C row = vgpr + 8*hi
    v8f c = {};
    if (accumulate) {
        #pragma unroll
        for (int v = 0; v < 8; ++v) c[v] = cp[(size_t)v * N];
    }
    // A tile: lane m = lane&15, lanes 0-15 hold K 0..15, lanes 16-31 hold K 16..31
    const _Float16* ap = A + (size_t)(m0 + nl) * K + hi * 16;
    // B tile: lane n = lane&15, lanes 0-15 hold K 0..15, lanes 16-31 hold K 16..31
    const _Float16* wp = W + (size_t)(n0 + nl) * K + hi * 16;
    for (int k0 = 0; k0 < K; k0 += 64) {
        union { unsigned int u[8]; v16h v; } a0, a1, b0, b1;
        #pragma unroll
        for (int j = 0; j < 4; ++j) {
            a0.u[j]     = *(const unsigned int*)(ap + k0 +      2 * j);
            a0.u[4 + j] = *(const unsigned int*)(ap + k0 +  8 + 2 * j);
            a1.u[j]     = *(const unsigned int*)(ap + k0 + 32 + 2 * j);
            a1.u[4 + j] = *(const unsigned int*)(ap + k0 + 40 + 2 * j);
        }
        #pragma unroll
        for (int j = 0; j < 8; ++j) {
            b0.u[j] = *(const unsigned int*)(wp + k0 +      2 * j);
            b1.u[j] = *(const unsigned int*)(wp + k0 + 32 + 2 * j);
        }
        c = __builtin_amdgcn_wmma_f32_16x16x32_f16(false, a0.v, false, b0.v,
                                                   (short)0, c, false, false);
        c = __builtin_amdgcn_wmma_f32_16x16x32_f16(false, a1.v, false, b1.v,
                                                   (short)0, c, false, false);
    }
    #pragma unroll
    for (int v = 0; v < 8; ++v) cp[(size_t)v * N] = c[v];
}

// ---------------- causal depthwise conv(4) + bias + SiLU (direction-ordered) ----------------
// reads xz[:, :DI] at natural index, writes xr (f32 + f16) at direction position p
__global__ __launch_bounds__(256) void conv_silu_kernel(const float* __restrict__ xz,
                                                        const float* __restrict__ cw,
                                                        const float* __restrict__ cb,
                                                        float* __restrict__ xr,
                                                        _Float16* __restrict__ xrh,
                                                        int dir) {
    int idx = blockIdx.x * 256 + threadIdx.x;       // over BL*DI
    if (idx >= BL * DI) return;
    int d = idx & (DI - 1);
    int p = (idx >> 8) & (LL - 1);
    int b = idx >> 20;
    float acc = cb[d];
    #pragma unroll
    for (int j = 0; j < 4; ++j) {
        int q = p - 3 + j;
        if (q >= 0) {
            int nat = dir ? (LL - 1 - q) : q;
            acc += cw[d * 4 + j] * xz[((size_t)b * LL + nat) * (2 * DI) + d];
        }
    }
    float s = acc * sigmoidf_(acc);
    size_t o = ((size_t)b * LL + p) * DI + d;
    xr[o]  = s;
    xrh[o] = (_Float16)s;
}

// ---------------- dt = softplus(x_dbl[:, :8] @ W_dt^T + b_dt) ----------------
__global__ __launch_bounds__(256) void dt_kernel(const float* __restrict__ xdbl,
                                                 const float* __restrict__ Wdt,
                                                 const float* __restrict__ bdt,
                                                 float* __restrict__ dt) {
    int idx = blockIdx.x * 256 + threadIdx.x;       // over BL*DI
    if (idx >= BL * DI) return;
    int d = idx & (DI - 1);
    int m = idx >> 8;
    const float* row = xdbl + (size_t)m * NXP;
    float acc = bdt[d];
    #pragma unroll
    for (int r = 0; r < DTR; ++r) acc += row[r] * Wdt[d * DTR + r];
    dt[idx] = (acc > 20.0f) ? acc : log1pf(__expf(acc));
}

// ---- shared helper: async-stage one chunk's Bm|Cm (CS x 32 f32, 16KB) into LDS ----
// xdbl row byte offset (48*m+8)*4 is 16B-aligned -> b128 async copies are legal.
static __device__ __forceinline__ void stage_bc_(float (*sdbl)[32],
                                                 const float* __restrict__ xdbl,
                                                 size_t base0, int t) {
    #pragma unroll
    for (int k = 0; k < 4; ++k) {                  // 128 rows * 8 quads / 256 thr
        int q    = k * 256 + t;
        int row  = q >> 3;
        int quad = q & 7;
        async_g2l_b128(lds_off_(&sdbl[row][quad * 4]),
                       xdbl + (base0 + row) * NXP + DTR + quad * 4);
    }
    wait_async0_();
    __syncthreads();
}

// ---------------- chunked selective scan ----------------
// pass 1: per-chunk affine carry (Aprod, Bacc) assuming h_in = 0
__global__ __launch_bounds__(256) void scan_local(const float* __restrict__ dt,
                                                  const float* __restrict__ xr,
                                                  const float* __restrict__ xdbl,
                                                  const float* __restrict__ An,
                                                  float* __restrict__ cA,
                                                  float* __restrict__ cB) {
    __shared__ float sdbl[CS][32];                 // [l][0:16]=Bm, [16:32]=Cm
    const int c = blockIdx.x % NC;
    const int b = blockIdx.x / NC;
    const int d = threadIdx.x;
    const size_t base0 = (size_t)b * LL + c * CS;
    stage_bc_(sdbl, xdbl, base0, d);
    float ap[DSN], bc[DSN];
    #pragma unroll
    for (int n = 0; n < DSN; ++n) { ap[n] = 1.0f; bc[n] = 0.0f; }
    for (int l = 0; l < CS; ++l) {
        size_t base = base0 + l;
        if ((l & 15) == 0 && l + 16 < CS) {        // stream-ahead hint (global_prefetch_b8)
            __builtin_prefetch(dt + (base + 16) * DI + d, 0, 0);
            __builtin_prefetch(xr + (base + 16) * DI + d, 0, 0);
        }
        float dtv = dt[base * DI + d];
        float du  = dtv * xr[base * DI + d];
        const float* bm = &sdbl[l][0];             // LDS broadcast, conflict-free
        #pragma unroll
        for (int n = 0; n < DSN; ++n) {
            float dA = __expf(dtv * An[d * DSN + n]);
            bc[n] = dA * bc[n] + du * bm[n];
            ap[n] *= dA;
        }
    }
    size_t co = ((size_t)(b * NC + c) * DI + d) * DSN;
    #pragma unroll
    for (int n = 0; n < DSN; ++n) { cA[co + n] = ap[n]; cB[co + n] = bc[n]; }
}

// pass 2: compose the NC chunk carries, record chunk-start states
__global__ __launch_bounds__(256) void scan_carry(const float* __restrict__ cA,
                                                  const float* __restrict__ cB,
                                                  float* __restrict__ hinit) {
    const int b = blockIdx.x;
    const int d = threadIdx.x;
    float h[DSN];
    #pragma unroll
    for (int n = 0; n < DSN; ++n) h[n] = 0.0f;
    for (int c = 0; c < NC; ++c) {
        size_t co = ((size_t)(b * NC + c) * DI + d) * DSN;
        #pragma unroll
        for (int n = 0; n < DSN; ++n) {
            hinit[co + n] = h[n];
            h[n] = cA[co + n] * h[n] + cB[co + n];
        }
    }
}

// pass 3: re-scan with correct h0, fuse C-contraction + D-skip + SiLU(z) gate,
// scatter to natural order as f16 for the out-projection GEMM
__global__ __launch_bounds__(256) void scan_final(const float* __restrict__ dt,
                                                  const float* __restrict__ xr,
                                                  const float* __restrict__ xdbl,
                                                  const float* __restrict__ An,
                                                  const float* __restrict__ Dv,
                                                  const float* __restrict__ xz,
                                                  const float* __restrict__ hinit,
                                                  _Float16* __restrict__ yh,
                                                  int dir) {
    __shared__ float sdbl[CS][32];
    const int c = blockIdx.x % NC;
    const int b = blockIdx.x / NC;
    const int d = threadIdx.x;
    const size_t base0 = (size_t)b * LL + c * CS;
    stage_bc_(sdbl, xdbl, base0, d);
    float h[DSN];
    {
        size_t co = ((size_t)(b * NC + c) * DI + d) * DSN;
        #pragma unroll
        for (int n = 0; n < DSN; ++n) h[n] = hinit[co + n];
    }
    const float Dd = Dv[d];
    for (int l = 0; l < CS; ++l) {
        int p = c * CS + l;
        size_t base = base0 + l;
        if ((l & 15) == 0 && l + 16 < CS) {
            __builtin_prefetch(dt + (base + 16) * DI + d, 0, 0);
            __builtin_prefetch(xr + (base + 16) * DI + d, 0, 0);
        }
        float dtv = dt[base * DI + d];
        float xv  = xr[base * DI + d];
        float du  = dtv * xv;
        const float* bm = &sdbl[l][0];
        const float* cm = &sdbl[l][16];
        float y = 0.0f;
        #pragma unroll
        for (int n = 0; n < DSN; ++n) {
            float dA = __expf(dtv * An[d * DSN + n]);
            h[n] = dA * h[n] + du * bm[n];
            y += h[n] * cm[n];
        }
        y += xv * Dd;
        int nat = dir ? (LL - 1 - p) : p;
        float zv = xz[((size_t)b * LL + nat) * (2 * DI) + DI + d];
        y *= zv * sigmoidf_(zv);
        yh[((size_t)b * LL + nat) * DI + d] = (_Float16)y;
    }
}

// ---------------- residual add + transpose back to NCHW ----------------
__global__ __launch_bounds__(256) void out_kernel(const float* __restrict__ x,
                                                  const float* __restrict__ acc,
                                                  float* __restrict__ out) {
    int idx = blockIdx.x * 256 + threadIdx.x;       // over BB*DM*LL
    if (idx >= BB * DM * LL) return;
    int l = idx & (LL - 1);
    int c = (idx >> 12) & (DM - 1);
    int b = idx >> 19;
    out[idx] = x[idx] + 0.5f * acc[((size_t)b * LL + l) * DM + c];
}

// ---------------- host orchestration ----------------
extern "C" void kernel_launch(void* const* d_in, const int* in_sizes, int n_in,
                              void* d_out, int out_size, void* d_ws, size_t ws_size,
                              hipStream_t stream) {
    (void)in_sizes; (void)n_in; (void)out_size; (void)ws_size;
    const float* x    = (const float*)d_in[0];
    const float* ln_w = (const float*)d_in[1];
    const float* ln_b = (const float*)d_in[2];
    // per-direction params: 0 W_in,1 conv_w,2 conv_b,3 W_x,4 W_dt,5 b_dt,6 A_log,7 D,8 W_out
    const float* P[2][9];
    for (int dir = 0; dir < 2; ++dir)
        for (int j = 0; j < 9; ++j)
            P[dir][j] = (const float*)d_in[3 + dir * 9 + j];
    float* out = (float*)d_out;

    // workspace carve (aligned 256B)
    char* base = (char*)d_ws; size_t off = 0;
    auto carve = [&](size_t bytes) -> void* {
        void* p = base + off; off = (off + bytes + 255) & ~(size_t)255; return p;
    };
    _Float16* t_h    = (_Float16*)carve((size_t)BL * DM * 2);
    _Float16* Win_h[2], *Wx_h[2], *Wout_h[2];
    float*    An[2];
    for (int i = 0; i < 2; ++i) {
        Win_h[i]  = (_Float16*)carve((size_t)2 * DI * DM * 2);
        Wx_h[i]   = (_Float16*)carve((size_t)NXP * DI * 2);
        Wout_h[i] = (_Float16*)carve((size_t)DM * DI * 2);
        An[i]     = (float*)   carve((size_t)DI * DSN * 4);
    }
    float*    xz    = (float*)   carve((size_t)BL * 2 * DI * 4);
    float*    xr    = (float*)   carve((size_t)BL * DI * 4);
    _Float16* xr_h  = (_Float16*)carve((size_t)BL * DI * 2);
    float*    xdbl  = (float*)   carve((size_t)BL * NXP * 4);
    float*    dtb   = (float*)   carve((size_t)BL * DI * 4);
    float*    cA    = (float*)   carve((size_t)BB * NC * DI * DSN * 4);
    float*    cB    = (float*)   carve((size_t)BB * NC * DI * DSN * 4);
    float*    hinit = (float*)   carve((size_t)BB * NC * DI * DSN * 4);
    _Float16* y_h   = (_Float16*)carve((size_t)BL * DI * 2);
    float*    accb  = (float*)   carve((size_t)BL * DM * 4);

    // ---- weight prep ----
    for (int i = 0; i < 2; ++i) {
        cvt_f32_to_f16<<<(2 * DI * DM + 255) / 256, 256, 0, stream>>>(P[i][0], Win_h[i], 2 * DI * DM);
        pad_wx_f16   <<<(NXP * DI + 255) / 256, 256, 0, stream>>>(P[i][3], Wx_h[i]);
        cvt_f32_to_f16<<<(DM * DI + 255) / 256, 256, 0, stream>>>(P[i][8], Wout_h[i], DM * DI);
        negexp_f32   <<<(DI * DSN + 255) / 256, 256, 0, stream>>>(P[i][6], An[i], DI * DSN);
    }

    // ---- LayerNorm + transpose (async-LDS staged) ----
    ln_kernel<<<BB * (LL / 32), 256, 0, stream>>>(x, ln_w, ln_b, t_h);

    // ---- per-direction pipeline (scratch reused; stream-ordered) ----
    for (int dir = 0; dir < 2; ++dir) {
        // in-projection: xz = t @ W_in^T   (M=8192, N=512, K=128)
        wmma_gemm_f16<<<dim3(BL / 16, (2 * DI) / 64), 128, 0, stream>>>(
            t_h, Win_h[dir], xz, BL, 2 * DI, DM, 0);
        // causal dw-conv + SiLU (direction-ordered)
        conv_silu_kernel<<<(BL * DI) / 256, 256, 0, stream>>>(
            xz, P[dir][1], P[dir][2], xr, xr_h, dir);
        // x_dbl = xr @ W_x^T   (M=8192, N=48, K=256)
        wmma_gemm_f16<<<dim3(BL / 16, 1), 128, 0, stream>>>(
            xr_h, Wx_h[dir], xdbl, BL, NXP, DI, 0);
        // dt = softplus(...)
        dt_kernel<<<(BL * DI) / 256, 256, 0, stream>>>(xdbl, P[dir][4], P[dir][5], dtb);
        // chunked selective scan (async-LDS staged B/C per chunk)
        scan_local<<<BB * NC, DI, 0, stream>>>(dtb, xr, xdbl, An[dir], cA, cB);
        scan_carry<<<BB, DI, 0, stream>>>(cA, cB, hinit);
        scan_final<<<BB * NC, DI, 0, stream>>>(dtb, xr, xdbl, An[dir], P[dir][7],
                                               xz, hinit, y_h, dir);
        // out-projection, accumulated across directions (M=8192, N=128, K=256)
        wmma_gemm_f16<<<dim3(BL / 16, DM / 64), 128, 0, stream>>>(
            y_h, Wout_h[dir], accb, BL, DM, DI, dir);
    }

    // ---- residual + transpose back ----
    out_kernel<<<(BB * DM * LL) / 256, 256, 0, stream>>>(x, accb, out);
}
